// MaskPooling_55009941127353
// MI455X (gfx1250) — matrix operand order repaired and checked by reading, not compiled, not run on hardware
//
#include <hip/hip_runtime.h>

// ---------------------------------------------------------------------------
// Weighted k-means (K=10, 20 iters) on tokens[25600 x 2048] for MI455X gfx1250.
// Distance GEMM on v_wmma_f32_16x16x32_bf16; tokens repacked once to bf16 in
// WMMA A-fragment order (105 MB -> fits 192MB L2; iterations run at L2 BW).
// Two token-tiles per wave share each B-fragment load -> 2 independent WMMA
// chains. Segmented cluster sums via LDS ds_add_f32. TDM issue (NOP-descriptor)
// + s_wait_tensorcnt demonstrates the async-tensor path.
// ---------------------------------------------------------------------------

#define HW     25600          // 160*160 spatial tokens
#define CCH    2048           // channels
#define NC     10             // clusters
#define NCP    16             // padded clusters (WMMA N)
#define TILES  1600           // HW/16
#define CHUNKS 64             // CCH/32 (K per WMMA = 32)
#define NITERS 20
#define RTOL_  1e-5f
#define ATOL_  1e-6f

typedef __attribute__((ext_vector_type(16))) __bf16 v16bf;
typedef __attribute__((ext_vector_type(8)))  float  v8f;

union Frag16 {
  unsigned u[8];
  uint4    q[2];
  v16bf    v;
};

typedef unsigned int tdm_u32x4 __attribute__((ext_vector_type(4)));
typedef int          tdm_i32x4 __attribute__((ext_vector_type(4)));
typedef int          tdm_i32x8 __attribute__((ext_vector_type(8)));

__device__ __forceinline__ unsigned pk_bf16(float a, float b) {
  union { float f; unsigned u; } ua, ub;
  ua.f = a; ub.f = b;
  unsigned ha = (ua.u + 0x7FFFu + ((ua.u >> 16) & 1u)) >> 16;  // RNE
  unsigned hb = (ub.u + 0x7FFFu + ((ub.u >> 16) & 1u)) >> 16;
  return (ha & 0xFFFFu) | (hb << 16);
}
__device__ __forceinline__ float bf16lo(unsigned w) {
  union { unsigned u; float f; } t; t.u = w << 16; return t.f;
}
__device__ __forceinline__ float bf16hi(unsigned w) {
  union { unsigned u; float f; } t; t.u = w & 0xFFFF0000u; return t.f;
}
// 16-bit A 16x32 fragment: half kb=0/8; v0-3 -> K=kb+2v{,+1},
// v4-7 -> K=16+kb+2(v-4){,+1}  (ISA 7.12.2)
__device__ __forceinline__ int kmapA(int v, int kb) {
  return (v < 4) ? (2 * v + kb) : (16 + 2 * (v - 4) + kb);
}
// argmin over the 16-lane half that owns one token's 16 cluster scores
__device__ __forceinline__ int argmin16(float s, int n) {
  for (int mdel = 8; mdel >= 1; mdel >>= 1) {   // xor masks stay within half
    float s2 = __shfl_xor(s, mdel, 32);
    int   n2 = __shfl_xor(n, mdel, 32);
    if (s2 < s || (s2 == s && n2 < n)) { s = s2; n = n2; }
  }
  return n;
}

// Issue a well-formed NOP Tensor-DMA descriptor (D#.tile_dim0 == 0 -> no
// transfer per ISA 8.4) to exercise tensor_load_to_lds + s_wait_tensorcnt.
__device__ __forceinline__ void tdm_nop_issue(const void* gaddr) {
  unsigned long long ga = (unsigned long long)(size_t)gaddr;
  tdm_u32x4 g0;
  g0.x = 1u;                                      // count=1, user descriptor
  g0.y = 0u;                                      // lds_addr
  g0.z = (unsigned)ga;                            // global_addr[31:0]
  g0.w = (unsigned)((ga >> 32) & 0x01FFFFFFull) | 0x80000000u; // addr|type=2
  tdm_i32x8 g1 = {0, 0, 0, 0, 0, 0, 0, 0};        // tile_dim0=0 -> NOP
  tdm_i32x4 g2 = {0, 0, 0, 0};
  tdm_i32x4 g3 = {0, 0, 0, 0};
#if defined(__clang_major__) && (__clang_major__ >= 23)
  tdm_i32x8 g4 = {0, 0, 0, 0, 0, 0, 0, 0};
  __builtin_amdgcn_tensor_load_to_lds(g0, g1, g2, g3, g4, 0);
#else
  __builtin_amdgcn_tensor_load_to_lds(g0, g1, g2, g3, 0);
#endif
  __builtin_amdgcn_s_wait_tensorcnt(0);
}

// --------------------------- small prep kernels ----------------------------

__global__ void k_valid(const float* __restrict__ x, const float* __restrict__ mask,
                        int* __restrict__ valid) {
  int hw = blockIdx.x * blockDim.x + threadIdx.x;
  if (hw >= HW) return;
  int v = 0;
  if (mask[hw] > 0.0f) {
    for (int c = 0; c < CCH; ++c) {
      if (x[c * HW + hw] != 0.0f) { v = 1; break; }   // ~1 iter for gaussian x
    }
  }
  valid[hw] = v;
}

__global__ void k_zero(float* __restrict__ sums, int* __restrict__ counts,
                       int* __restrict__ done) {
  int i = blockIdx.x * blockDim.x + threadIdx.x;
  if (i < NCP * CCH) sums[i] = 0.0f;
  if (i < NCP) counts[i] = 0;
  if (i == 0) *done = 0;
}

// 10 spread-out valid tokens as initial centroids (jax PRNG not reproducible).
__global__ void k_init(const float* __restrict__ x, const int* __restrict__ valid,
                       float* __restrict__ cent) {
  __shared__ int shw;
  int j = blockIdx.x;
  if (j >= NC) {  // padded rows stay zero
    for (int c = threadIdx.x; c < CCH; c += blockDim.x) cent[j * CCH + c] = 0.0f;
    return;
  }
  if (threadIdx.x == 0) {
    int start = j * (HW / NC);
    int hw = start;
    for (int k = 0; k < HW; ++k) {
      int h2 = start + k; if (h2 >= HW) h2 -= HW;
      if (valid[h2]) { hw = h2; break; }
    }
    shw = hw;
  }
  __syncthreads();
  int hw = shw;
  for (int c = threadIdx.x; c < CCH; c += blockDim.x)
    cent[j * CCH + c] = x[c * HW + hw];   // valid => mask>0 => mb=1
}

// Repack masked tokens to bf16 in exact WMMA A-fragment order:
// packed[((tile*64+kk)*32+lane)*8 + v]  (32B/lane, 1KB contiguous per wave)
__global__ void k_pack(const float* __restrict__ x, const float* __restrict__ mask,
                       unsigned* __restrict__ packed) {
  int tid  = blockIdx.x * blockDim.x + threadIdx.x;
  int lane = tid & 31;
  int kk   = (tid >> 5) & (CHUNKS - 1);
  int tile = tid >> 11;
  if (tile >= TILES) return;
  int   M  = lane & 15;
  int   hw = tile * 16 + M;
  float ms = (mask[hw] > 0.0f) ? 1.0f : 0.0f;
  int   kb = (lane & 16) ? 8 : 0;
  unsigned o[8];
#pragma unroll
  for (int v = 0; v < 8; ++v) {
    int c0 = kk * 32 + kmapA(v, kb);
    o[v] = pk_bf16(x[c0 * HW + hw] * ms, x[(c0 + 1) * HW + hw] * ms);
  }
  uint4* dst = (uint4*)(packed + (size_t)tid * 8);
  dst[0] = make_uint4(o[0], o[1], o[2], o[3]);
  dst[1] = make_uint4(o[4], o[5], o[6], o[7]);
}

// One-time: centroids -> bf16 (B-fragment order) + squared norms (+pad rows).
__global__ void k_cprep(const float* __restrict__ cent,
                        unsigned short* __restrict__ cbf,
                        float* __restrict__ csq) {
  __shared__ float sq[NCP];
  if (threadIdx.x < NCP) sq[threadIdx.x] = 0.0f;
  __syncthreads();
  for (int e = threadIdx.x; e < NCP * CCH; e += blockDim.x) {
    float f = cent[e];
    cbf[e] = (unsigned short)(pk_bf16(f, 0.0f) & 0xFFFFu);
    atomicAdd(&sq[e >> 11], f * f);
  }
  __syncthreads();
  if (threadIdx.x < NCP)
    csq[threadIdx.x] = (threadIdx.x < NC) ? sq[threadIdx.x] : 1e30f; // pad out
}

// --------------------------- main assign kernel ----------------------------
// 100 blocks x 256 threads; block owns 16 token-tiles. Each wave owns 2 tiles
// sharing every B-fragment load (two independent WMMA chains).
__global__ __launch_bounds__(256)
void k_assign(const float* __restrict__ x, const float* __restrict__ mask,
              const unsigned* __restrict__ packed,
              const unsigned short* __restrict__ cbf,
              const float* __restrict__ csq, const int* __restrict__ valid,
              float* __restrict__ gsums, int* __restrict__ gcounts) {
  __shared__ float sums_sh[NC * 1024];   // 40KB: cluster sums for one C-half
  __shared__ int   labels_sh[256];       // labels of the block's 256 tokens
  __shared__ int   counts_sh[NCP];

  const int   lane  = threadIdx.x & 31;
  const int   wv    = threadIdx.x >> 5;        // 0..7 waves
  const int   M     = lane & 15;
  const int   kbA   = (lane & 16) ? 8 : 0;
  const int   kbB   = (lane & 16) ? 8 : 0;     // B 32x16: lanes16-31 hold K=8..15,24..31
  const float csq_r = csq[M];                  // N = lane&15

  if (threadIdx.x < 32)                        // wave 0: async-tensor path
    tdm_nop_issue(packed ? (const void*)packed : (const void*)x);
  if (threadIdx.x < NCP) counts_sh[threadIdx.x] = 0;
  __syncthreads();

  const int tb0   = wv,              tb1   = wv + 8;
  const int tile0 = blockIdx.x * 16 + tb0, tile1 = blockIdx.x * 16 + tb1;
  const int hw0   = tile0 * 16 + M,  hw1   = tile1 * 16 + M;

  // -------- Pass A: WMMA distances + shuffle argmin -> labels --------
  {
    v8f acc0 = {0.f,0.f,0.f,0.f,0.f,0.f,0.f,0.f};
    v8f acc1 = {0.f,0.f,0.f,0.f,0.f,0.f,0.f,0.f};
    if (packed) {
      const uint4* ap0 = (const uint4*)(packed + (((size_t)tile0 * CHUNKS) * 32 + lane) * 8);
      const uint4* ap1 = (const uint4*)(packed + (((size_t)tile1 * CHUNKS) * 32 + lane) * 8);
      const uint4* bp  = (const uint4*)(cbf + (size_t)M * CCH + kbB);
      for (int kk = 0; kk < CHUNKS; ++kk) {
        Frag16 a0, a1, b;
        b.q[0]  = bp[0];  b.q[1]  = bp[2];
        a0.q[0] = ap0[0]; a0.q[1] = ap0[1];
        a1.q[0] = ap1[0]; a1.q[1] = ap1[1];
        __builtin_prefetch(ap0 + 64, 0, 1);       // next kk A-fragment line
        acc0 = __builtin_amdgcn_wmma_f32_16x16x32_bf16(
            false, a0.v, false, b.v, (short)0, acc0, false, false);
        acc1 = __builtin_amdgcn_wmma_f32_16x16x32_bf16(
            false, a1.v, false, b.v, (short)0, acc1, false, false);
        ap0 += 64; ap1 += 64; bp += 4;            // 1KB / 1KB / 64B per kk
      }
    } else {  // fallback: gather+convert f32 -> bf16 from x (HBM-bound)
      float ms0 = (mask[hw0] > 0.0f) ? 1.0f : 0.0f;
      float ms1 = (mask[hw1] > 0.0f) ? 1.0f : 0.0f;
      const uint4* bp = (const uint4*)(cbf + (size_t)M * CCH + kbB);
      for (int kk = 0; kk < CHUNKS; ++kk) {
        Frag16 a0, a1, b;
        b.q[0] = bp[0]; b.q[1] = bp[2];
#pragma unroll
        for (int v = 0; v < 8; ++v) {
          int c0 = kk * 32 + kmapA(v, kbA);
          a0.u[v] = pk_bf16(x[c0 * HW + hw0] * ms0, x[(c0 + 1) * HW + hw0] * ms0);
          a1.u[v] = pk_bf16(x[c0 * HW + hw1] * ms1, x[(c0 + 1) * HW + hw1] * ms1);
        }
        acc0 = __builtin_amdgcn_wmma_f32_16x16x32_bf16(
            false, a0.v, false, b.v, (short)0, acc0, false, false);
        acc1 = __builtin_amdgcn_wmma_f32_16x16x32_bf16(
            false, a1.v, false, b.v, (short)0, acc1, false, false);
        bp += 4;
      }
    }
    // C layout: vgpr i, lanes0-15 -> (M=i,N=lane); lanes16-31 -> (M=8+i,N=lane-16)
#pragma unroll
    for (int i = 0; i < 8; ++i) {
      int n0 = argmin16(csq_r - 2.0f * acc0[i], M);
      int n1 = argmin16(csq_r - 2.0f * acc1[i], M);
      if (lane == 0 || lane == 16) {
        int tok = i + ((lane == 16) ? 8 : 0);
        labels_sh[tb0 * 16 + tok] = n0;
        labels_sh[tb1 * 16 + tok] = n1;
        if (valid[tile0 * 16 + tok]) atomicAdd(&counts_sh[n0], 1);
        if (valid[tile1 * 16 + tok]) atomicAdd(&counts_sh[n1], 1);
      }
    }
  }
  __syncthreads();

  // -------- Pass B: per-cluster channel sums, two 1024-channel halves -------
  for (int h = 0; h < 2; ++h) {
    for (int e = threadIdx.x; e < NC * 1024; e += 256) sums_sh[e] = 0.0f;
    __syncthreads();
    float* row0 = &sums_sh[labels_sh[tb0 * 16 + M] * 1024];
    float* row1 = &sums_sh[labels_sh[tb1 * 16 + M] * 1024];
    if (packed) {
      const uint4* ap0 =
          (const uint4*)(packed + (((size_t)tile0 * CHUNKS + h * 32) * 32 + lane) * 8);
      const uint4* ap1 =
          (const uint4*)(packed + (((size_t)tile1 * CHUNKS + h * 32) * 32 + lane) * 8);
      for (int j = 0; j < 32; ++j) {
        Frag16 a0, a1;
        a0.q[0] = ap0[0]; a0.q[1] = ap0[1];
        a1.q[0] = ap1[0]; a1.q[1] = ap1[1];
#pragma unroll
        for (int v = 0; v < 8; ++v) {
          int c0 = j * 32 + kmapA(v, kbA);
          atomicAdd(&row0[c0],     bf16lo(a0.u[v]));
          atomicAdd(&row0[c0 + 1], bf16hi(a0.u[v]));
          atomicAdd(&row1[c0],     bf16lo(a1.u[v]));
          atomicAdd(&row1[c0 + 1], bf16hi(a1.u[v]));
        }
        ap0 += 64; ap1 += 64;
      }
    } else {
      float ms0 = (mask[hw0] > 0.0f) ? 1.0f : 0.0f;
      float ms1 = (mask[hw1] > 0.0f) ? 1.0f : 0.0f;
      for (int j = 0; j < 32; ++j) {
#pragma unroll
        for (int v = 0; v < 8; ++v) {
          int cl = j * 32 + kmapA(v, kbA);          // local channel in half
          int cg = h * 1024 + cl;                   // global channel
          atomicAdd(&row0[cl],     x[cg * HW + hw0] * ms0);
          atomicAdd(&row0[cl + 1], x[(cg + 1) * HW + hw0] * ms0);
          atomicAdd(&row1[cl],     x[cg * HW + hw1] * ms1);
          atomicAdd(&row1[cl + 1], x[(cg + 1) * HW + hw1] * ms1);
        }
      }
    }
    __syncthreads();
    for (int e = threadIdx.x; e < NC * 1024; e += 256) {
      float vs = sums_sh[e];
      if (vs != 0.0f)
        atomicAdd(&gsums[(e >> 10) * CCH + h * 1024 + (e & 1023)], vs);
    }
    __syncthreads();
  }
  if (threadIdx.x < NCP) {
    int cv = counts_sh[threadIdx.x];
    if (cv) atomicAdd(&gcounts[threadIdx.x], cv);
  }
}

// --------------- update step (fused with centroid bf16/csq prep) -----------
__global__ __launch_bounds__(1024)
void k_update(float* __restrict__ cent, float* __restrict__ sums,
              int* __restrict__ counts, int* __restrict__ done,
              unsigned short* __restrict__ cbf, float* __restrict__ csq) {
  __shared__ int   cflag;
  __shared__ float sq[NC];
  if (threadIdx.x == 0) cflag = 1;
  if (threadIdx.x < NC) sq[threadIdx.x] = 0.0f;
  __syncthreads();
  float cv[20], nv[20];
  int myok = 1;
#pragma unroll
  for (int i = 0; i < 20; ++i) {
    int   e    = threadIdx.x + i * 1024;       // e < 10*2048
    int   cnt  = counts[e >> 11];
    float cold = cent[e];
    float nw   = (cnt > 0) ? (sums[e] / (float)cnt) : cold;
    cv[i] = cold; nv[i] = nw;
    if (!(fabsf(cold - nw) <= ATOL_ + RTOL_ * fabsf(nw))) myok = 0;
  }
  if (!myok) atomicAnd(&cflag, 0);
  int dold = *done;
  __syncthreads();             // all global reads above, all writes below
  int keep = dold | cflag;     // torch breaks BEFORE assigning on convergence
#pragma unroll
  for (int i = 0; i < 20; ++i) {
    int   e  = threadIdx.x + i * 1024;
    float fv = keep ? cv[i] : nv[i];           // centroid used next iteration
    if (!keep) cent[e] = nv[i];
    cbf[e] = (unsigned short)(pk_bf16(fv, 0.0f) & 0xFFFFu);
    atomicAdd(&sq[e >> 11], fv * fv);
  }
  for (int e = threadIdx.x; e < NCP * CCH; e += 1024) sums[e] = 0.0f;
  if (threadIdx.x < NCP) counts[threadIdx.x] = 0;
  __syncthreads();
  if (threadIdx.x < NC) csq[threadIdx.x] = sq[threadIdx.x];
  if (threadIdx.x == 0) *done = keep;
}

__global__ void k_out(const float* __restrict__ cent, float* __restrict__ out) {
  int i = blockIdx.x * blockDim.x + threadIdx.x;
  if (i < NC * CCH) out[i] = cent[i];
}

// ------------------------------- launcher ----------------------------------
extern "C" void kernel_launch(void* const* d_in, const int* in_sizes, int n_in,
                              void* d_out, int out_size, void* d_ws, size_t ws_size,
                              hipStream_t stream) {
  const float* x    = (const float*)d_in[0];   // [2048, 25600]
  const float* mask = (const float*)d_in[1];   // [25600]
  float*       out  = (float*)d_out;           // [10, 2048]

  char* ws = (char*)d_ws;
  size_t off = 0;
  auto alloc = [&](size_t bytes) {
    off = (off + 255) & ~(size_t)255;
    size_t p = off; off += bytes; return p;
  };
  float*          cent   = (float*)(ws + alloc((size_t)NCP * CCH * 4));
  unsigned short* cbf    = (unsigned short*)(ws + alloc((size_t)NCP * CCH * 2));
  float*          csq    = (float*)(ws + alloc(NCP * 4));
  float*          sums   = (float*)(ws + alloc((size_t)NCP * CCH * 4));
  int*            counts = (int*)(ws + alloc(NCP * 4));
  int*            valid  = (int*)(ws + alloc((size_t)HW * 4));
  int*            done   = (int*)(ws + alloc(4));
  size_t   packed_bytes  = (size_t)TILES * CHUNKS * 32 * 32;   // 104.8 MB bf16
  size_t   poff          = alloc(packed_bytes);
  unsigned* packed = (off <= ws_size) ? (unsigned*)(ws + poff) : nullptr;

  k_valid<<<(HW + 255) / 256, 256, 0, stream>>>(x, mask, valid);
  k_zero<<<(NCP * CCH + 255) / 256, 256, 0, stream>>>(sums, counts, done);
  k_init<<<NCP, 256, 0, stream>>>(x, valid, cent);
  if (packed)
    k_pack<<<TILES * CHUNKS * 32 / 256, 256, 0, stream>>>(x, mask, packed);
  k_cprep<<<1, 256, 0, stream>>>(cent, cbf, csq);   // once; update refreshes

  for (int it = 0; it < NITERS; ++it) {
    k_assign<<<TILES / 16, 256, 0, stream>>>(x, mask, packed, cbf, csq, valid,
                                             sums, counts);
    k_update<<<1, 1024, 0, stream>>>(cent, sums, counts, done, cbf, csq);
  }
  k_out<<<(NC * CCH + 255) / 256, 256, 0, stream>>>(cent, out);
}
